// HardestContrastiveLoss_70995809402955
// MI455X (gfx1250) — compile-verified, matrix operand order checked by reading, not run on hardware
//
#include <hip/hip_runtime.h>
#include <hip/hip_bf16.h>
#include <math.h>

// ---------------- problem constants (from reference) ----------------
#define NUM_NODE   8192
#define N_PTS      20000
#define FEAT_DIM   32
#define POS_THRESH 0.1f
#define NEG_THRESH 1.4f
#define EPS_F      1e-7f
// reference tests sqrt(max(d2,0)+eps) vs radius -> equivalent to d2 vs r^2-eps
#define POS_R2E    (0.0375f * 0.0375f - EPS_F)
#define NEG_R2E    (0.1f * 0.1f - EPS_F)
#define BIG_F      100000.0f
#define SENT_NEG   (-1.0e30f)   // "no positive seen" sentinel (d2 domain, max)
#define SENT_POS   ( 1.0e30f)   // "no negative seen" sentinel (d2 domain, min)

#define COL_CHUNKS 4            // column split for occupancy: 2048 waves total
#define NTILE_ALL  (NUM_NODE / 16)          // 512 column tiles
#define NTILE_CHK  (NTILE_ALL / COL_CHUNKS) // 128 tiles per chunk

typedef __attribute__((ext_vector_type(16))) _Float16 v16h;
typedef __attribute__((ext_vector_type(8)))  float    v8f;

// ---------------------------------------------------------------------------
// Stage 1: gather + transform + pack feats into WMMA A/B lane layout.
// Packed row layout (32 halves): [K0..7 | K16..23 | K8..15 | K24..31]
//   -> lanes 0-15 read halves [0..15], lanes 16-31 read halves [16..31],
//      matching the 16-bit 16x32 A-matrix (and symmetric B) VGPR layout.
// ---------------------------------------------------------------------------
__global__ void __launch_bounds__(256)
hcl_prep_kernel(const float* __restrict__ src_pcd,
                const float* __restrict__ tgt_pcd,
                const float* __restrict__ src_feats,
                const float* __restrict__ tgt_feats,
                const int*   __restrict__ corr,
                const float* __restrict__ rot,
                const float* __restrict__ trans,
                float4* __restrict__ sp4, float4* __restrict__ tp4,
                _Float16* __restrict__ aPack, _Float16* __restrict__ bPack,
                float* __restrict__ na, float* __restrict__ nb)
{
    int t = blockIdx.x * blockDim.x + threadIdx.x;
    if (t >= NUM_NODE) return;

    int ci = corr[2 * t + 0];
    int cj = corr[2 * t + 1];

    // src point through rot/trans
    float px = src_pcd[3 * ci + 0];
    float py = src_pcd[3 * ci + 1];
    float pz = src_pcd[3 * ci + 2];
    float sx = rot[0] * px + rot[1] * py + rot[2] * pz + trans[0];
    float sy = rot[3] * px + rot[4] * py + rot[5] * pz + trans[1];
    float sz = rot[6] * px + rot[7] * py + rot[8] * pz + trans[2];
    sp4[t] = make_float4(sx, sy, sz, 0.0f);
    tp4[t] = make_float4(tgt_pcd[3 * cj + 0], tgt_pcd[3 * cj + 1],
                         tgt_pcd[3 * cj + 2], 0.0f);

    float accA = 0.0f, accB = 0.0f;
#pragma unroll
    for (int d = 0; d < FEAT_DIM; ++d) {
        int g = d >> 3;                               // dest group of 8
        int k = (((g & 1) << 4) | ((g >> 1) << 3)) + (d & 7); // {0,16,8,24}+..
        _Float16 ha = (_Float16)src_feats[(size_t)ci * FEAT_DIM + k];
        _Float16 hb = (_Float16)tgt_feats[(size_t)cj * FEAT_DIM + k];
        aPack[(size_t)t * FEAT_DIM + d] = ha;
        bPack[(size_t)t * FEAT_DIM + d] = hb;
        float fa = (float)ha, fb = (float)hb;
        accA += fa * fa;
        accB += fb * fb;
    }
    na[t] = accA;
    nb[t] = accB;
}

// ---------------------------------------------------------------------------
// Stage 2: fused distance matrices + row max/min in squared-distance domain
// (sqrt deferred to stage 3; masked entries use +/- sentinels).
// One wave32 per (16-row strip x column chunk): 512 strips x 4 chunks =
// 2048 waves. Per tile: 1x v_wmma_f32_16x16x32_f16 -> 16x16 dot products.
// A-tile is loaded once per wave and reused across all 128 column tiles.
// ---------------------------------------------------------------------------
__global__ void __launch_bounds__(256)
hcl_tile_kernel(const _Float16* __restrict__ aPack,
                const _Float16* __restrict__ bPack,
                const float*    __restrict__ na,
                const float*    __restrict__ nb,
                const float4*   __restrict__ sp4,
                const float4*   __restrict__ tp4,
                float* __restrict__ fpOut,   // [COL_CHUNKS][NUM_NODE] partial max
                float* __restrict__ cnOut)   // [COL_CHUNKS][NUM_NODE] partial min
{
    const int lane    = threadIdx.x & 31;
    const int gw      = blockIdx.x * (blockDim.x >> 5) + (threadIdx.x >> 5);
    const int chunk   = gw / (NUM_NODE / 16);     // column chunk [0,4)
    const int strip   = gw % (NUM_NODE / 16);     // row strip    [0,512)
    const int rowBase = strip * 16;
    const int tile0   = chunk * NTILE_CHK;
    const int l15     = lane & 15;
    const int hi      = lane >> 4;   // 0: lanes 0-15, 1: lanes 16-31

    // A tile: fixed for the whole strip. Lane covers row rowBase+l15,
    // K-half selected by `hi` thanks to the pre-swizzled packing.
    v16h a = *(const v16h*)(aPack + ((size_t)(rowBase + l15) * FEAT_DIM) + hi * 16);

    // C/D layout: lane holds rows (r + 8*hi), column l15.
    float  naR[8];
    float4 spR[8];
#pragma unroll
    for (int r = 0; r < 8; ++r) {
        int row = rowBase + r + hi * 8;
        naR[r] = na[row];
        spR[r] = sp4[row];
    }

    float rowmax[8], rowmin[8];
#pragma unroll
    for (int r = 0; r < 8; ++r) { rowmax[r] = SENT_NEG; rowmin[r] = SENT_POS; }

    for (int tile = tile0; tile < tile0 + NTILE_CHK; ++tile) {
        const int j = tile * 16 + l15;

        v16h   b   = *(const v16h*)(bPack + ((size_t)j * FEAT_DIM) + hi * 16);
        float  nbj = nb[j];
        float4 tp  = tp4[j];

        if (tile + 1 < tile0 + NTILE_CHK)  // pull next B tile into WGP cache
            __builtin_prefetch(bPack + ((size_t)(j + 16) * FEAT_DIM), 0, 3);

        v8f c = {};
        // 8 args: (neg_a, A, neg_b, B, c_mod, C, reuse_a, reuse_b)
        c = __builtin_amdgcn_wmma_f32_16x16x32_f16(
                false, a, false, b, (short)0, c, false, false);

#pragma unroll
        for (int r = 0; r < 8; ++r) {
            // feature distance^2: |a|^2 + |b|^2 - 2 a.b  (no sqrt here)
            float d2f = fmaf(-2.0f, c[r], naR[r] + nbj);
            // point distance^2 vs (radius^2 - eps): sqrt is monotone
            float dx  = spR[r].x - tp.x;
            float dy  = spR[r].y - tp.y;
            float dz  = spR[r].z - tp.z;
            float d2p = fmaf(dx, dx, fmaf(dy, dy, dz * dz));
            rowmax[r] = fmaxf(rowmax[r], (d2p < POS_R2E) ? d2f : SENT_NEG);
            rowmin[r] = fminf(rowmin[r], (d2p > NEG_R2E) ? d2f : SENT_POS);
        }
    }

    // Cross-lane reduction within each group of 16 lanes (rows are disjoint
    // between the two halves, so width-16 xor shuffles reduce both at once).
#pragma unroll
    for (int r = 0; r < 8; ++r) {
        float mx = rowmax[r];
        float mn = rowmin[r];
#pragma unroll
        for (int m = 8; m >= 1; m >>= 1) {
            mx = fmaxf(mx, __shfl_xor(mx, m, 16));
            mn = fminf(mn, __shfl_xor(mn, m, 16));
        }
        if (l15 == 0) {
            int row = rowBase + r + hi * 8;
            fpOut[(size_t)chunk * NUM_NODE + row] = mx;
            cnOut[(size_t)chunk * NUM_NODE + row] = mn;
        }
    }
}

// ---------------------------------------------------------------------------
// Stage 3: fold the 4 column-chunk partials, apply the deferred sqrt
// (8192 of them -> negligible), then
// loss = mean(relu(fp - 0.1)) + mean(relu(1.4 - cn))
// ---------------------------------------------------------------------------
__global__ void __launch_bounds__(256)
hcl_reduce_kernel(const float* __restrict__ fp,
                  const float* __restrict__ cn,
                  float* __restrict__ out)
{
    __shared__ float sP[256];
    __shared__ float sN[256];
    float p = 0.0f, n = 0.0f;
    for (int i = threadIdx.x; i < NUM_NODE; i += 256) {
        float mx = SENT_NEG;
        float mn = SENT_POS;
#pragma unroll
        for (int c = 0; c < COL_CHUNKS; ++c) {
            mx = fmaxf(mx, fp[(size_t)c * NUM_NODE + i]);
            mn = fminf(mn, cn[(size_t)c * NUM_NODE + i]);
        }
        // no positive -> furthest_positive = 0 (max of zeroed row)
        float fd = (mx > -1.0e29f)
                       ? __builtin_amdgcn_sqrtf(fmaxf(mx, 0.0f) + EPS_F)
                       : 0.0f;
        // no negative -> closest_negative = BIG
        float cd = (mn < 1.0e29f)
                       ? __builtin_amdgcn_sqrtf(fmaxf(mn, 0.0f) + EPS_F)
                       : BIG_F;
        p += fmaxf(fd - POS_THRESH, 0.0f);
        n += fmaxf(NEG_THRESH - cd, 0.0f);
    }
    sP[threadIdx.x] = p;
    sN[threadIdx.x] = n;
    __syncthreads();
    for (int s = 128; s > 0; s >>= 1) {
        if (threadIdx.x < s) {
            sP[threadIdx.x] += sP[threadIdx.x + s];
            sN[threadIdx.x] += sN[threadIdx.x + s];
        }
        __syncthreads();
    }
    if (threadIdx.x == 0)
        out[0] = (sP[0] + sN[0]) * (1.0f / (float)NUM_NODE);
}

// ---------------------------------------------------------------------------
extern "C" void kernel_launch(void* const* d_in, const int* in_sizes, int n_in,
                              void* d_out, int out_size, void* d_ws, size_t ws_size,
                              hipStream_t stream)
{
    const float* src_pcd   = (const float*)d_in[0];
    const float* tgt_pcd   = (const float*)d_in[1];
    const float* src_feats = (const float*)d_in[2];
    const float* tgt_feats = (const float*)d_in[3];
    const int*   corr      = (const int*)  d_in[4];
    const float* rot       = (const float*)d_in[5];
    const float* trans     = (const float*)d_in[6];
    float* out = (float*)d_out;

    // workspace carve-up (all offsets 256B-aligned)
    char* ws = (char*)d_ws;
    size_t off = 0;
    float4*   sp4   = (float4*)  (ws + off); off += (size_t)NUM_NODE * sizeof(float4);
    float4*   tp4   = (float4*)  (ws + off); off += (size_t)NUM_NODE * sizeof(float4);
    _Float16* aPack = (_Float16*)(ws + off); off += (size_t)NUM_NODE * FEAT_DIM * sizeof(_Float16);
    _Float16* bPack = (_Float16*)(ws + off); off += (size_t)NUM_NODE * FEAT_DIM * sizeof(_Float16);
    float*    na    = (float*)   (ws + off); off += (size_t)NUM_NODE * sizeof(float);
    float*    nb    = (float*)   (ws + off); off += (size_t)NUM_NODE * sizeof(float);
    float*    fpA   = (float*)   (ws + off); off += (size_t)COL_CHUNKS * NUM_NODE * sizeof(float);
    float*    cnA   = (float*)   (ws + off); off += (size_t)COL_CHUNKS * NUM_NODE * sizeof(float);

    // Stage 1: gather/transform/pack
    hcl_prep_kernel<<<NUM_NODE / 256, 256, 0, stream>>>(
        src_pcd, tgt_pcd, src_feats, tgt_feats, corr, rot, trans,
        sp4, tp4, aPack, bPack, na, nb);

    // Stage 2: 512 strips x 4 column chunks = 2048 waves, 8 waves/block
    hcl_tile_kernel<<<(NTILE_ALL * COL_CHUNKS) / 8, 256, 0, stream>>>(
        aPack, bPack, na, nb, sp4, tp4, fpA, cnA);

    // Stage 3: final scalar
    hcl_reduce_kernel<<<1, 256, 0, stream>>>(fpA, cnA, out);
}
